// TransformerBlock_44152263803529
// MI455X (gfx1250) — matrix operand order repaired and checked by reading, not compiled, not run on hardware
//
#include <hip/hip_runtime.h>
#include <hip/hip_bf16.h>

typedef __bf16 bf16_t;
typedef __attribute__((ext_vector_type(16))) __bf16 v16bf;
typedef __attribute__((ext_vector_type(8)))  float  v8f;
typedef __attribute__((ext_vector_type(4)))  unsigned int v4u;
typedef __attribute__((ext_vector_type(8)))  int v8i;
typedef __attribute__((ext_vector_type(4)))  int v4i;

#define B_   2
#define S_   2048
#define D_   2048
#define KV_  256
#define H_   32
#define HD_  64
#define DFF_ 5632
#define BS_  (B_*S_)
#define EPS_ 1e-6f
#define SCALE_ (1.0f/(16.0f+1e-6f))

union FragU { v16bf v; uint4 q[2]; };

// A-matrix 16x32 bf16 frag: lane<16 -> row, K {0..7,16..23}; lane>=16 -> K {8..15,24..31}
__device__ __forceinline__ v16bf ldsFragA(const bf16_t* base, int ld, int row0, int k0, int lane) {
    int r  = row0 + (lane & 15);
    int kb = (lane >> 4) << 3;            // 0 or 8
    FragU f;
    f.q[0] = *(const uint4*)(base + r * ld + k0 + kb);
    f.q[1] = *(const uint4*)(base + r * ld + k0 + kb + 16);
    return f.v;
}
// B-matrix 32x16 bf16 frag: lane<16 -> col, K {0..7,16..23}; lane>=16 -> K {8..15,24..31}
__device__ __forceinline__ v16bf ldsFragB(const bf16_t* base, int ld, int row0, int k0, int lane) {
    int r  = row0 + (lane & 15);
    int kb = (lane >> 4) << 3;            // 0 or 8
    FragU f;
    f.q[0] = *(const uint4*)(base + r * ld + k0 + kb);
    f.q[1] = *(const uint4*)(base + r * ld + k0 + kb + 16);
    return f.v;
}
__device__ __forceinline__ v8f wmma_bf16(v16bf a, v16bf b, v8f c) {
    return __builtin_amdgcn_wmma_f32_16x16x32_bf16(false, a, false, b, (short)0, c, false, false);
}

// ---- Tensor Data Mover: DMA a 2D bf16 tile [tileY rows x tileX elems] from
// global (row stride = strideElems) into LDS at ldsAddr. D# per ISA 8.3/8.4.
// 6-arg builtin form (clang-23 / therock-10.0 headers).
__device__ __forceinline__ void tdm_load_2d(unsigned int ldsAddr, const bf16_t* g,
                                            unsigned int tileX, unsigned int tileY,
                                            unsigned int strideElems) {
    unsigned long long ga = (unsigned long long)(size_t)g;
    v4u g0;
    g0.x = 1u;                                                      // count=1, user descriptor
    g0.y = ldsAddr;                                                 // lds_addr (bytes)
    g0.z = (unsigned int)ga;                                        // global_addr[31:0]
    g0.w = (unsigned int)((ga >> 32) & 0x01FFFFFFu) | (2u << 30);   // global_addr[56:32] | type=2
    unsigned int td0 = strideElems;        // tensor_dim0 >= tileX (tile fully in-bounds)
    unsigned int td1 = 1u << 30;           // tensor_dim1: effectively unbounded
    v8i g1;
    g1[0] = (int)0x00010000u;                       // data_size = 2 bytes; no pad/iterate/multicast
    g1[1] = (int)(td0 << 16);                       // tensor_dim0[15:0]
    g1[2] = (int)((td0 >> 16) | (td1 << 16));       // tensor_dim0[31:16] | tensor_dim1[15:0]
    g1[3] = (int)((td1 >> 16) | (tileX << 16));     // tensor_dim1[31:16] | tile_dim0
    g1[4] = (int)(tileY & 0xFFFFu);                 // tile_dim1 | tile_dim2=0
    g1[5] = (int)strideElems;                       // tensor_dim0_stride[31:0]
    g1[6] = 0;                                      // stride[47:32] | tensor_dim1_stride[15:0]
    g1[7] = 0;
    v4i z4; z4[0] = z4[1] = z4[2] = z4[3] = 0;
    v8i z8;
    for (int i = 0; i < 8; ++i) z8[i] = 0;
    __builtin_amdgcn_tensor_load_to_lds(g0, g1, z4, z4, z8, 0);
}

// ---------------- elementwise helpers ----------------

__global__ void f32_to_bf16_kernel(const float* __restrict__ in, bf16_t* __restrict__ out, long n) {
    long i = (long)blockIdx.x * blockDim.x + threadIdx.x;
    long stride = (long)gridDim.x * blockDim.x;
    for (; i < n; i += stride) out[i] = (bf16_t)in[i];
}

__global__ __launch_bounds__(256) void rmsnorm_kernel(const float* __restrict__ x,
                                                      const float* __restrict__ w,
                                                      bf16_t* __restrict__ out) {
    __shared__ float red[256];
    const int tid = threadIdx.x;
    const long row = blockIdx.x;
    const float* xr = x + row * (long)D_;
    float ss = 0.f;
    for (int i = tid; i < D_; i += 256) { float v = xr[i]; ss += v * v; }
    red[tid] = ss;
    __syncthreads();
    for (int s2 = 128; s2 > 0; s2 >>= 1) {
        if (tid < s2) red[tid] += red[tid + s2];
        __syncthreads();
    }
    float inv = 1.f / (sqrtf(red[0] * (1.f / D_) + EPS_) + EPS_);
    bf16_t* orow = out + row * (long)D_;
    for (int i = tid; i < D_; i += 256) orow[i] = (bf16_t)(xr[i] * inv * w[i]);
}

// out[i] = rotate_half(t)[i] * (pm0[s,i] + pm1[s,i])   (faithful to reference)
__global__ void posemb_kernel(const float* __restrict__ t, const float* __restrict__ pm,
                              bf16_t* __restrict__ out, int dim, long total) {
    long idx = (long)blockIdx.x * blockDim.x + threadIdx.x;
    if (idx >= total) return;
    long row = idx / dim;
    int  i   = (int)(idx - row * dim);
    int  s   = (int)(row & (S_ - 1));
    int  hd  = dim >> 1;
    float src = (i < hd) ? -t[row * (long)dim + i + hd] : t[row * (long)dim + i - hd];
    float f = pm[(long)s * D_ + i] + pm[(long)S_ * D_ + (long)s * D_ + i];
    out[idx] = (bf16_t)(src * f);
}

// ---------------- WMMA GEMM:  out[M,N] = A[M,K] @ W[N,K]^T ----------------
// MODE 0: outF = A@W^T ; MODE 1: outF = resid + A@W^T ; MODE 2: outB = bf16(A@W^T)
// TDM double-buffered staging: wave 0 issues tensor_load_to_lds for tile kt+1
// while all 8 waves compute on tile kt.

template <int MODE>
__global__ __launch_bounds__(256) void gemm_kernel(const bf16_t* __restrict__ A,
                                                   const bf16_t* __restrict__ W,
                                                   const float* __restrict__ resid,
                                                   float* __restrict__ outF,
                                                   bf16_t* __restrict__ outB,
                                                   int M, int N, int K) {
    __shared__ alignas(16) bf16_t sA[2][128 * 32];
    __shared__ alignas(16) bf16_t sW[2][128 * 32];
    const int tid = threadIdx.x, lane = tid & 31, wid = tid >> 5;
    const int wm = wid >> 1, wn = wid & 1;
    const long bM = (long)blockIdx.y * 128, bN = (long)blockIdx.x * 128;

    v8f acc[2][4];
    for (int s = 0; s < 2; ++s)
        for (int n = 0; n < 4; ++n)
            for (int i = 0; i < 8; ++i) acc[s][n][i] = 0.f;

    const int nk = K / 32;
    if (wid == 0) {
        tdm_load_2d((unsigned int)(size_t)&sA[0][0], A + bM * K, 32, 128, (unsigned int)K);
        tdm_load_2d((unsigned int)(size_t)&sW[0][0], W + bN * K, 32, 128, (unsigned int)K);
    }
    int buf = 0;
    for (int kt = 0; kt < nk; ++kt) {
        if (wid == 0) {
            if (kt + 1 < nk) {
                tdm_load_2d((unsigned int)(size_t)&sA[buf ^ 1][0], A + bM * K + (kt + 1) * 32, 32, 128, (unsigned int)K);
                tdm_load_2d((unsigned int)(size_t)&sW[buf ^ 1][0], W + bN * K + (kt + 1) * 32, 32, 128, (unsigned int)K);
                __builtin_amdgcn_s_wait_tensorcnt((short)2);   // tile kt landed; kt+1 may be in flight
            } else {
                __builtin_amdgcn_s_wait_tensorcnt((short)0);
            }
        }
        __syncthreads();
        v16bf af[2], wf[4];
        for (int s = 0; s < 2; ++s) af[s] = ldsFragA(sA[buf], 32, wm * 32 + s * 16, 0, lane);
        for (int n = 0; n < 4; ++n) wf[n] = ldsFragB(sW[buf], 32, wn * 64 + n * 16, 0, lane);
        for (int s = 0; s < 2; ++s)
            for (int n = 0; n < 4; ++n) acc[s][n] = wmma_bf16(af[s], wf[n], acc[s][n]);
        __syncthreads();                                       // all reads done before DMA overwrites
        buf ^= 1;
    }
    const int half = lane >> 4, lc = lane & 15;
    for (int s = 0; s < 2; ++s)
        for (int n = 0; n < 4; ++n)
            for (int i = 0; i < 8; ++i) {
                long row = bM + wm * 32 + s * 16 + half * 8 + i;
                long col = bN + wn * 64 + n * 16 + lc;
                long o = row * (long)N + col;
                float v = acc[s][n][i];
                if (MODE == 1) v += resid[o];
                if (MODE == 2) outB[o] = (bf16_t)v;
                else           outF[o] = v;
            }
}

// ---------------- fused gate/up GEMM:  out = bf16( silu(A@Wg^T) * (A@Wu^T) ) ----------------
// Block tile 128x64, each wave 32x32 per matrix (64 accum VGPRs) to avoid spills.

__global__ __launch_bounds__(256) void gemm_gateup_kernel(const bf16_t* __restrict__ A,
                                                          const bf16_t* __restrict__ Wg,
                                                          const bf16_t* __restrict__ Wu,
                                                          bf16_t* __restrict__ out,
                                                          int M, int N, int K) {
    __shared__ alignas(16) bf16_t sA[2][128 * 32];
    __shared__ alignas(16) bf16_t sG[2][64 * 32];
    __shared__ alignas(16) bf16_t sU[2][64 * 32];
    const int tid = threadIdx.x, lane = tid & 31, wid = tid >> 5;
    const int wm = wid >> 1, wn = wid & 1;
    const long bM = (long)blockIdx.y * 128, bN = (long)blockIdx.x * 64;

    v8f accg[2][2], accu[2][2];
    for (int s = 0; s < 2; ++s)
        for (int n = 0; n < 2; ++n)
            for (int i = 0; i < 8; ++i) { accg[s][n][i] = 0.f; accu[s][n][i] = 0.f; }

    const int nk = K / 32;
    if (wid == 0) {
        tdm_load_2d((unsigned int)(size_t)&sA[0][0], A  + bM * K, 32, 128, (unsigned int)K);
        tdm_load_2d((unsigned int)(size_t)&sG[0][0], Wg + bN * K, 32,  64, (unsigned int)K);
        tdm_load_2d((unsigned int)(size_t)&sU[0][0], Wu + bN * K, 32,  64, (unsigned int)K);
    }
    int buf = 0;
    for (int kt = 0; kt < nk; ++kt) {
        if (wid == 0) {
            if (kt + 1 < nk) {
                tdm_load_2d((unsigned int)(size_t)&sA[buf ^ 1][0], A  + bM * K + (kt + 1) * 32, 32, 128, (unsigned int)K);
                tdm_load_2d((unsigned int)(size_t)&sG[buf ^ 1][0], Wg + bN * K + (kt + 1) * 32, 32,  64, (unsigned int)K);
                tdm_load_2d((unsigned int)(size_t)&sU[buf ^ 1][0], Wu + bN * K + (kt + 1) * 32, 32,  64, (unsigned int)K);
                __builtin_amdgcn_s_wait_tensorcnt((short)3);
            } else {
                __builtin_amdgcn_s_wait_tensorcnt((short)0);
            }
        }
        __syncthreads();
        v16bf af[2];
        for (int s = 0; s < 2; ++s) af[s] = ldsFragA(sA[buf], 32, wm * 32 + s * 16, 0, lane);
        for (int n = 0; n < 2; ++n) {
            v16bf gf = ldsFragB(sG[buf], 32, wn * 32 + n * 16, 0, lane);
            v16bf uf = ldsFragB(sU[buf], 32, wn * 32 + n * 16, 0, lane);
            for (int s = 0; s < 2; ++s) {
                accg[s][n] = wmma_bf16(af[s], gf, accg[s][n]);
                accu[s][n] = wmma_bf16(af[s], uf, accu[s][n]);
            }
        }
        __syncthreads();
        buf ^= 1;
    }
    const int half = lane >> 4, lc = lane & 15;
    for (int s = 0; s < 2; ++s)
        for (int n = 0; n < 2; ++n)
            for (int i = 0; i < 8; ++i) {
                long row = bM + wm * 32 + s * 16 + half * 8 + i;
                long col = bN + wn * 32 + n * 16 + lc;
                float g = accg[s][n][i];
                float u = accu[s][n][i];
                float sg = g / (1.f + __expf(-g));
                out[row * (long)N + col] = (bf16_t)(sg * u);
            }
}

// ---------------- flash-style attention ----------------
// score = (Q K^T) * SCALE ; attn = e^s / (sum e^s + 1) ; out = attn @ V
// head h uses kv columns [(h&3)*64, +64) of the shared KV projection.

__global__ __launch_bounds__(128) void attn_kernel(const bf16_t* __restrict__ q,
                                                   const bf16_t* __restrict__ k,
                                                   const bf16_t* __restrict__ v,
                                                   bf16_t* __restrict__ o) {
    __shared__ alignas(16) bf16_t sK[64 * 64];
    __shared__ alignas(16) bf16_t sVt[64 * 64];
    __shared__ alignas(16) bf16_t sP[4 * 16 * 64];
    const int tid = threadIdx.x, lane = tid & 31, wid = tid >> 5;
    const int lc = lane & 15, half = lane >> 4;
    const int b = blockIdx.z, h = blockIdx.y, qt = blockIdx.x;
    const long qrow0 = (long)b * S_ + qt * 64 + wid * 16;
    const int kvoff = (h & 3) * HD_;

    // Q A-frags (16 rows x 64 HD), loaded straight from global in frag layout
    v16bf aq[2];
    for (int ks = 0; ks < 2; ++ks) {
        const bf16_t* g = q + (qrow0 + lc) * (long)D_ + h * HD_ + ks * 32 + half * 8;
        FragU f;
        f.q[0] = *(const uint4*)g;
        f.q[1] = *(const uint4*)(g + 16);
        aq[ks] = f.v;
    }
    float mrun[8], lrun[8];
    v8f oacc[4];
    for (int i = 0; i < 8; ++i) { mrun[i] = -3.0e38f; lrun[i] = 0.f; }
    for (int n = 0; n < 4; ++n)
        for (int i = 0; i < 8; ++i) oacc[n][i] = 0.f;

    for (int kt = 0; kt < S_ / 64; ++kt) {
        __syncthreads();
        const long krow0 = (long)b * S_ + kt * 64;
        if (wid == 0)                                  // DMA K tile (64x64, stride KV)
            tdm_load_2d((unsigned int)(size_t)sK, k + krow0 * KV_ + kvoff, 64, 64, (unsigned int)KV_);
        for (int e = 0; e < 32; ++e) {                 // V staged transposed: sVt[d][k]
            int idx = e * 128 + tid;
            int kr = idx >> 6, d = idx & 63;
            sVt[d * 64 + kr] = v[(krow0 + kr) * (long)KV_ + kvoff + d];
        }
        if (wid == 0) __builtin_amdgcn_s_wait_tensorcnt((short)0);
        __syncthreads();

        // scores: Q[16x64] @ K[64x64]^T
        v8f sc[4];
        for (int n = 0; n < 4; ++n)
            for (int i = 0; i < 8; ++i) sc[n][i] = 0.f;
        for (int ks = 0; ks < 2; ++ks)
            for (int n = 0; n < 4; ++n) {
                v16bf bk = ldsFragB(sK, 64, n * 16, ks * 32, lane);
                sc[n] = wmma_bf16(aq[ks], bk, sc[n]);
            }

        // online softmax stats (row-wise over 16 lanes of each half)
        float mt[8];
        for (int i = 0; i < 8; ++i) {
            float m = sc[0][i] * SCALE_;
            for (int n = 1; n < 4; ++n) m = fmaxf(m, sc[n][i] * SCALE_);
            for (int off = 1; off < 16; off <<= 1) m = fmaxf(m, __shfl_xor(m, off, 32));
            mt[i] = m;
        }
        float alpha[8], rs[8];
        for (int i = 0; i < 8; ++i) {
            float mn = fmaxf(mrun[i], mt[i]);
            alpha[i] = __expf(mrun[i] - mn);
            mrun[i] = mn;
            rs[i] = 0.f;
        }
        for (int n = 0; n < 4; ++n)
            for (int i = 0; i < 8; ++i) {
                float p = __expf(sc[n][i] * SCALE_ - mrun[i]);
                rs[i] += p;
                sP[wid * 1024 + (half * 8 + i) * 64 + n * 16 + lc] = (bf16_t)p;
            }
        for (int i = 0; i < 8; ++i) {
            float r = rs[i];
            for (int off = 1; off < 16; off <<= 1) r += __shfl_xor(r, off, 32);
            lrun[i] = lrun[i] * alpha[i] + r;
        }
        for (int n = 0; n < 4; ++n)
            for (int i = 0; i < 8; ++i) oacc[n][i] *= alpha[i];

        // P[16x64] @ V[64x64]
        for (int ks = 0; ks < 2; ++ks) {
            v16bf ap = ldsFragA(sP + wid * 1024, 64, 0, ks * 32, lane);
            for (int n = 0; n < 4; ++n) {
                v16bf bv = ldsFragB(sVt, 64, n * 16, ks * 32, lane);
                oacc[n] = wmma_bf16(ap, bv, oacc[n]);
            }
        }
    }
    // denominator = sum e^s + 1 = e^m * (l + e^-m)
    for (int i = 0; i < 8; ++i) lrun[i] = lrun[i] + __expf(-mrun[i]);
    for (int n = 0; n < 4; ++n)
        for (int i = 0; i < 8; ++i) {
            long row = qrow0 + half * 8 + i;
            long col = h * HD_ + n * 16 + lc;
            o[row * (long)D_ + col] = (bf16_t)(oacc[n][i] / lrun[i]);
        }
}

// ---------------- host orchestration ----------------

extern "C" void kernel_launch(void* const* d_in, const int* in_sizes, int n_in,
                              void* d_out, int out_size, void* d_ws, size_t ws_size,
                              hipStream_t stream) {
    (void)in_sizes; (void)n_in; (void)out_size; (void)ws_size;
    const float* x    = (const float*)d_in[0];
    const float* pm   = (const float*)d_in[1];
    const float* nw_a = (const float*)d_in[2];
    const float* nw_f = (const float*)d_in[3];
    const float* wq   = (const float*)d_in[4];
    const float* wk   = (const float*)d_in[5];
    const float* wv   = (const float*)d_in[6];
    const float* wo   = (const float*)d_in[7];
    const float* wg   = (const float*)d_in[8];
    const float* wu   = (const float*)d_in[9];
    const float* wd   = (const float*)d_in[10];
    float* out = (float*)d_out;

    char* ws = (char*)d_ws;
    size_t cur = 0;
    auto alloc = [&](size_t bytes) -> void* {
        cur = (cur + 255) & ~(size_t)255;
        void* p = ws + cur;
        cur += bytes;
        return p;
    };
    bf16_t* wq_b  = (bf16_t*)alloc((size_t)D_ * D_ * 2);
    bf16_t* wk_b  = (bf16_t*)alloc((size_t)KV_ * D_ * 2);
    bf16_t* wv_b  = (bf16_t*)alloc((size_t)KV_ * D_ * 2);
    bf16_t* wo_b  = (bf16_t*)alloc((size_t)D_ * D_ * 2);
    bf16_t* wg_b  = (bf16_t*)alloc((size_t)DFF_ * D_ * 2);
    bf16_t* wu_b  = (bf16_t*)alloc((size_t)DFF_ * D_ * 2);
    bf16_t* wd_b  = (bf16_t*)alloc((size_t)D_ * DFF_ * 2);
    bf16_t* h_b   = (bf16_t*)alloc((size_t)BS_ * D_ * 2);
    float*  qtmp  = (float*) alloc((size_t)BS_ * D_ * 4);
    float*  ktmp  = (float*) alloc((size_t)BS_ * KV_ * 4);
    bf16_t* q_b   = (bf16_t*)alloc((size_t)BS_ * D_ * 2);
    bf16_t* k_b   = (bf16_t*)alloc((size_t)BS_ * KV_ * 2);
    bf16_t* v_b   = (bf16_t*)alloc((size_t)BS_ * KV_ * 2);
    bf16_t* ao_b  = (bf16_t*)alloc((size_t)BS_ * D_ * 2);
    bf16_t* h2_b  = (bf16_t*)alloc((size_t)BS_ * D_ * 2);
    bf16_t* ff_b  = (bf16_t*)alloc((size_t)BS_ * DFF_ * 2);

    // weights -> bf16
    f32_to_bf16_kernel<<<2048, 256, 0, stream>>>(wq, wq_b, (long)D_ * D_);
    f32_to_bf16_kernel<<<1024, 256, 0, stream>>>(wk, wk_b, (long)KV_ * D_);
    f32_to_bf16_kernel<<<1024, 256, 0, stream>>>(wv, wv_b, (long)KV_ * D_);
    f32_to_bf16_kernel<<<2048, 256, 0, stream>>>(wo, wo_b, (long)D_ * D_);
    f32_to_bf16_kernel<<<2048, 256, 0, stream>>>(wg, wg_b, (long)DFF_ * D_);
    f32_to_bf16_kernel<<<2048, 256, 0, stream>>>(wu, wu_b, (long)DFF_ * D_);
    f32_to_bf16_kernel<<<2048, 256, 0, stream>>>(wd, wd_b, (long)D_ * DFF_);

    // attention sublayer
    rmsnorm_kernel<<<BS_, 256, 0, stream>>>(x, nw_a, h_b);
    gemm_kernel<0><<<dim3(D_ / 128, BS_ / 128), 256, 0, stream>>>(h_b, wq_b, nullptr, qtmp, nullptr, BS_, D_, D_);
    gemm_kernel<0><<<dim3(KV_ / 128, BS_ / 128), 256, 0, stream>>>(h_b, wk_b, nullptr, ktmp, nullptr, BS_, KV_, D_);
    gemm_kernel<2><<<dim3(KV_ / 128, BS_ / 128), 256, 0, stream>>>(h_b, wv_b, nullptr, nullptr, v_b, BS_, KV_, D_);
    posemb_kernel<<<((long)BS_ * D_) / 256, 256, 0, stream>>>(qtmp, pm, q_b, D_, (long)BS_ * D_);
    posemb_kernel<<<((long)BS_ * KV_) / 256, 256, 0, stream>>>(ktmp, pm, k_b, KV_, (long)BS_ * KV_);
    attn_kernel<<<dim3(S_ / 64, H_, B_), 128, 0, stream>>>(q_b, k_b, v_b, ao_b);
    gemm_kernel<1><<<dim3(D_ / 128, BS_ / 128), 256, 0, stream>>>(ao_b, wo_b, x, out, nullptr, BS_, D_, D_);

    // feedforward sublayer
    rmsnorm_kernel<<<BS_, 256, 0, stream>>>(out, nw_f, h2_b);
    gemm_gateup_kernel<<<dim3(DFF_ / 64, BS_ / 128), 256, 0, stream>>>(h2_b, wg_b, wu_b, ff_b, BS_, DFF_, D_);
    gemm_kernel<1><<<dim3(D_ / 128, BS_ / 128), 256, 0, stream>>>(ff_b, wd_b, out, out, nullptr, BS_, D_, DFF_);
}
